// DirectedDAGNN_86225763434541
// MI455X (gfx1250) — matrix operand order repaired and checked by reading, not compiled, not run on hardware
//
#include <hip/hip_runtime.h>
#include <hip/hip_bf16.h>

// ---------------------------------------------------------------------------
// DAGNN on MI455X (gfx1250, wave32).
//   - Dense layers + head: v_wmma_f32_16x16x32_f16, fully unrolled K loop
//     (compile-time K/NCOLS/flags), one wave per 16x16 output tile.
//   - APPNP propagation: L2-resident SpMM via global_atomic_add_f32, 0.9
//     folded into edge_norm, 0.1*h0 seeded into the destination buffer.
// ---------------------------------------------------------------------------

typedef __attribute__((ext_vector_type(16))) _Float16 v16h;
typedef __attribute__((ext_vector_type(8)))  float    v8f;

#define NN   50000
#define EE   800000
#define INF  256
#define HH   128
#define HH2  64
#define OO   32
#define KHOP 6            // K+1 attention weights
#define EPSB 1e-5f

// ---------------- simple elementwise kernels ----------------

__global__ void zero_f32(float* __restrict__ p, int n) {
    int i = blockIdx.x * blockDim.x + threadIdx.x;
    if (i < n) p[i] = 0.0f;
}

__global__ void wsum_scatter(const int* __restrict__ row, const float* __restrict__ w,
                             float* __restrict__ wsum, int e_cnt) {
    int e = blockIdx.x * blockDim.x + threadIdx.x;
    if (e < e_cnt) atomicAdd(&wsum[row[e]], w[e]);
}

__global__ void edge_norm_k(const int* __restrict__ row, const float* __restrict__ w,
                            const float* __restrict__ wsum, float* __restrict__ norm9,
                            int e_cnt) {
    int e = blockIdx.x * blockDim.x + threadIdx.x;
    if (e < e_cnt) norm9[e] = 0.9f * w[e] / fmaxf(wsum[row[e]], 1.0f);
}

// next = ALPHA * h0 (float4-wide; also the zero-init of the scatter target)
__global__ void seed_blend(const float4* __restrict__ h0, float4* __restrict__ nxt, int n4) {
    int i = blockIdx.x * blockDim.x + threadIdx.x;
    if (i < n4) {
        float4 v = h0[i];
        nxt[i] = make_float4(0.1f * v.x, 0.1f * v.y, 0.1f * v.z, 0.1f * v.w);
    }
}

// fused (+)= softmax(att)[k] * src ; softmax over 6 scalars computed inline
__global__ void fused_axpy(const float* __restrict__ att, int k,
                           const float4* __restrict__ src, float4* __restrict__ fused,
                           int n4, int init) {
    int i = blockIdx.x * blockDim.x + threadIdx.x;
    if (i >= n4) return;
    float mx = att[0];
    #pragma unroll
    for (int j = 1; j < KHOP; ++j) mx = fmaxf(mx, att[j]);
    float s = 0.0f, num = 0.0f;
    #pragma unroll
    for (int j = 0; j < KHOP; ++j) {
        float e = __expf(att[j] - mx);
        s += e;
        if (j == k) num = e;
    }
    const float wk = num / s;
    float4 v = src[i];
    float4 acc = init ? make_float4(0.f, 0.f, 0.f, 0.f) : fused[i];
    acc.x += wk * v.x; acc.y += wk * v.y; acc.z += wk * v.z; acc.w += wk * v.w;
    fused[i] = acc;
}

// ---------------- SpMM scatter: one wave per edge, float4 per lane ----------

__global__ void spmm_scatter(const int* __restrict__ row, const int* __restrict__ col,
                             const float* __restrict__ norm9,
                             const float* __restrict__ cur, float* __restrict__ nxt,
                             int e_cnt) {
    const int lane = threadIdx.x & 31;
    const int wave = threadIdx.x >> 5;
    const int e = blockIdx.x * 8 + wave;
    if (e >= e_cnt) return;
    const int r = row[e];
    const int c = col[e];
    const float w = norm9[e];
    const float4 v = ((const float4*)(cur + (long long)r * HH))[lane];
    float* dst = nxt + (long long)c * HH + lane * 4;
    atomicAdd(dst + 0, w * v.x);   // global_atomic_add_f32 (no return)
    atomicAdd(dst + 1, w * v.y);
    atomicAdd(dst + 2, w * v.z);
    atomicAdd(dst + 3, w * v.w);
}

// ---------------- fused WMMA GEMM: out = post(A @ W + bias) ----------------
// A: [n_rows, K] f32 row-major, W: [K, NCOLS] f32 row-major.
// One wave computes one 16x16 output tile. K, NCOLS, and the epilogue flags
// are compile-time so the K loop fully unrolls into back-to-back v_wmma ops.

template <int K, int NCOLS, bool BN, bool RELU, bool RES>
__global__ void wmma_gemm_fused(const float* __restrict__ A, const float* __restrict__ W,
                                const float* __restrict__ bias,
                                const float* __restrict__ gamma, const float* __restrict__ beta,
                                const float* __restrict__ mean,  const float* __restrict__ var,
                                const float* residual,
                                float* __restrict__ out, int n_rows) {
    constexpr int TILES_X = NCOLS / 16;
    const int lane  = threadIdx.x & 31;
    const int wave  = threadIdx.x >> 5;
    const int tile  = blockIdx.x * 8 + wave;
    const int total = (n_rows >> 4) * TILES_X;
    if (tile >= total) return;                  // whole-wave exit, EXEC stays full

    const int tm    = tile / TILES_X;
    const int tn    = tile % TILES_X;
    const int half  = lane >> 4;                // K-half split for A fragment
    const int m     = tm * 16 + (lane & 15);
    const int cbase = tn * 16;

    v8f c = {};
    #pragma unroll
    for (int k0 = 0; k0 < K; k0 += 32) {
        v16h a, b;
        // A fragment: lane holds row m; per-VGPR K mapping per ISA 16-bit A
        // table. K-pairs are contiguous -> float2 loads.
        const float* arow = A + (long long)m * K + k0;
        #pragma unroll
        for (int v = 0; v < 8; ++v) {
            const int kk = ((v >> 2) << 4) + (half << 3) + ((v & 3) << 1);
            const float2 p = *(const float2*)(arow + kk);
            a[2 * v]     = (_Float16)p.x;
            a[2 * v + 1] = (_Float16)p.y;
        }
        // B fragment: lane holds column N=cbase+(lane&15); K mapping matches A
        // (ISA 16-bit table order), scalar strided loads.
        const float* wcol = W + (long long)k0 * NCOLS + cbase + (lane & 15);
        #pragma unroll
        for (int v = 0; v < 8; ++v) {
            const int kk = ((v >> 2) << 4) + (half << 3) + ((v & 3) << 1);
            b[2 * v]     = (_Float16)wcol[(long long)kk * NCOLS];
            b[2 * v + 1] = (_Float16)wcol[(long long)(kk + 1) * NCOLS];
        }
        c = __builtin_amdgcn_wmma_f32_16x16x32_f16(
                /*neg_a=*/false, a, /*neg_b=*/false, b,
                /*c_mod=*/(short)0, c, /*reuse_a=*/false, /*reuse_b=*/false);
    }

    // Epilogue: coli is invariant across the 8 accumulator rows, so fold
    // bias + BN into one fma (val*scale + shift) computed once per wave.
    const int coli = cbase + (lane & 15);
    float scale = 1.0f, shift = bias[coli];
    if (BN) {
        const float sc = gamma[coli] * rsqrtf(var[coli] + EPSB);
        shift = (shift - mean[coli]) * sc + beta[coli];
        scale = sc;
    }
    #pragma unroll
    for (int r = 0; r < 8; ++r) {
        const int rowi = tm * 16 + half * 8 + r;
        float val = c[r] * scale + shift;
        if (RELU) val = fmaxf(val, 0.0f);
        if (RES)  val += residual[(long long)rowi * NCOLS + coli];
        out[(long long)rowi * NCOLS + coli] = val;
    }
}

// ---------------------------------------------------------------------------

extern "C" void kernel_launch(void* const* d_in, const int* in_sizes, int n_in,
                              void* d_out, int out_size, void* d_ws, size_t ws_size,
                              hipStream_t stream) {
    const float* x   = (const float*)d_in[0];
    const int*   ei  = (const int*)  d_in[1];
    const float* ew  = (const float*)d_in[2];
    const float* W1  = (const float*)d_in[3];
    const float* b1  = (const float*)d_in[4];
    const float* g1  = (const float*)d_in[5];
    const float* be1 = (const float*)d_in[6];
    const float* m1  = (const float*)d_in[7];
    const float* v1  = (const float*)d_in[8];
    const float* W2  = (const float*)d_in[9];
    const float* b2  = (const float*)d_in[10];
    const float* g2  = (const float*)d_in[11];
    const float* be2 = (const float*)d_in[12];
    const float* m2  = (const float*)d_in[13];
    const float* v2  = (const float*)d_in[14];
    const float* att = (const float*)d_in[15];
    const float* W3  = (const float*)d_in[16];
    const float* b3  = (const float*)d_in[17];
    const float* g3  = (const float*)d_in[18];
    const float* be3 = (const float*)d_in[19];
    const float* m3  = (const float*)d_in[20];
    const float* v3  = (const float*)d_in[21];
    const float* W4  = (const float*)d_in[22];
    const float* b4  = (const float*)d_in[23];
    float* out = (float*)d_out;

    const int* row = ei;          // edge_index[0]
    const int* col = ei + EE;     // edge_index[1]

    // Workspace layout (floats). Total ~26.45M floats (~106 MB) — all feature
    // buffers are L2-resident on MI455X (192 MB L2).
    float* ws    = (float*)d_ws;
    float* wsum  = ws;                              // NN
    float* norm9 = ws + NN;                         // EE   (0.9 * edge_norm)
    float* buf0  = ws + NN + EE;                    // NN*HH
    float* buf1  = buf0 + (size_t)NN * HH;          // NN*HH
    float* buf2  = buf1 + (size_t)NN * HH;          // NN*HH
    float* fused = buf2 + (size_t)NN * HH;          // NN*HH

    const int nNH = NN * HH;
    const int n4  = nNH / 4;

    // --- weighted edge normalization ---
    zero_f32    <<<(NN + 255) / 256, 256, 0, stream>>>(wsum, NN);
    wsum_scatter<<<(EE + 255) / 256, 256, 0, stream>>>(row, ew, wsum, EE);
    edge_norm_k <<<(EE + 255) / 256, 256, 0, stream>>>(row, ew, wsum, norm9, EE);

    // --- lin1 + bn1 + relu -> buf0 ---
    wmma_gemm_fused<INF, HH, true, true, false>
        <<<(NN / 16) * (HH / 16) / 8, 256, 0, stream>>>(
            x, W1, b1, g1, be1, m1, v1, nullptr, buf0, NN);

    // --- lin2 + bn2 + relu + residual -> buf1 (xs[0] / initial hc) ---
    wmma_gemm_fused<HH, HH, true, true, true>
        <<<(NN / 16) * (HH / 16) / 8, 256, 0, stream>>>(
            buf0, W2, b2, g2, be2, m2, v2, buf0, buf1, NN);

    float* h0 = buf1;
    float* sA = buf2;
    float* sB = buf0;

    // fused = w0 * h0
    fused_axpy<<<(n4 + 255) / 256, 256, 0, stream>>>(att, 0, (const float4*)h0,
                                                     (float4*)fused, n4, 1);

    for (int k = 0; k < 5; ++k) {                   // outer hops
        float* cur   = h0;
        float* dst   = sA;
        float* other = sB;
        for (int i = 0; i < 5; ++i) {               // inner APPNP iterations
            seed_blend  <<<(n4 + 255) / 256, 256, 0, stream>>>(
                (const float4*)h0, (float4*)dst, n4);
            spmm_scatter<<<(EE + 7) / 8, 256, 0, stream>>>(row, col, norm9, cur, dst, EE);
            cur = dst;
            float* t = dst; dst = other; other = t;
        }
        // cur holds prop(h0); rotate buffers so old h0 becomes scratch
        float* s1 = h0;
        float* s2 = (cur == sA) ? sB : sA;
        h0 = cur; sA = s1; sB = s2;
        fused_axpy<<<(n4 + 255) / 256, 256, 0, stream>>>(att, k + 1, (const float4*)h0,
                                                         (float4*)fused, n4, 0);
    }

    // --- head: Linear(H,H/2)+BN+ReLU -> sA ; Linear(H/2,O) -> out ---
    const int tiles_h1 = (NN / 16) * (HH2 / 16);
    wmma_gemm_fused<HH, HH2, true, true, false>
        <<<(tiles_h1 + 7) / 8, 256, 0, stream>>>(
            fused, W3, b3, g3, be3, m3, v3, nullptr, sA, NN);

    const int tiles_h2 = (NN / 16) * (OO / 16);
    wmma_gemm_fused<HH2, OO, false, false, false>
        <<<(tiles_h2 + 7) / 8, 256, 0, stream>>>(
            sA, W4, b4, nullptr, nullptr, nullptr, nullptr, nullptr, out, NN);
}
